// BiasMultiHeadAttention_19069654794762
// MI455X (gfx1250) — matrix operand order repaired and checked by reading, not compiled, not run on hardware
//
#include <hip/hip_runtime.h>

// ---------------------------------------------------------------------------
// BiasMultiHeadAttention for MI455X (gfx1250, wave32, WMMA).
// B=2, N=512, D=256, H=8, hd=32.
// Dominant cost: edge MLP over E [2,512,512,256] f32 (537 MB HBM stream @
// 23.3 TB/s ~= 23 us floor; 68.7 GFLOP GEMM) -> bf16 WMMA, We1 fragments
// staged in LDS per workgroup, each B-frag reused for 2 WMMAs (2x M-tiling),
// E loads non-temporal, relu + [256,1] projection fused into the epilogue.
// ---------------------------------------------------------------------------

typedef __attribute__((ext_vector_type(16))) __bf16 v16bf;
typedef __attribute__((ext_vector_type(8)))  __bf16 v8bf;
typedef __attribute__((ext_vector_type(8)))  float  v8f;
typedef __attribute__((ext_vector_type(4)))  float  v4f;   // native clang vector

#define WMMA_BF16(A, B, C) \
  __builtin_amdgcn_wmma_f32_16x16x32_bf16(false, (A), false, (B), (short)0, (C), false, false)

static __device__ __forceinline__ v8f vzero8() {
  v8f z;
#pragma unroll
  for (int i = 0; i < 8; ++i) z[i] = 0.f;
  return z;
}

// Load a 16x32 bf16 A-fragment from a row-major f32 matrix (non-temporal:
// used for the single-pass E / X / O streams).
// A k order used here (contiguous, matches B packing): lanes 0-15: M=lane, K = 0..15;
// lanes 16-31: M=lane-16, K = 16..31.
static __device__ __forceinline__ v16bf load_a_f32_nt(const float* base_row, int kofs, int lane) {
  const int base = (lane >= 16) ? 16 : 0;
  const v4f* p0 = (const v4f*)(base_row + kofs + base);
  const v4f* p1 = (const v4f*)(base_row + kofs + 8 + base);
  v4f a0 = __builtin_nontemporal_load(p0);
  v4f a1 = __builtin_nontemporal_load(p0 + 1);
  v4f b0 = __builtin_nontemporal_load(p1);
  v4f b1 = __builtin_nontemporal_load(p1 + 1);
  v16bf r;
#pragma unroll
  for (int e = 0; e < 4; ++e) {
    r[e]      = (__bf16)a0[e];
    r[4 + e]  = (__bf16)a1[e];
    r[8 + e]  = (__bf16)b0[e];
    r[12 + e] = (__bf16)b1[e];
  }
  return r;
}

// ---------------------------------------------------------------------------
// Kernel 1: repack 256x256 f32 weights into bf16 WMMA-B fragment order.
// B k order used here (contiguous, matches A loads): lanes 0-15: N=lane, K=e (0..15);
// lanes 16-31: N=lane-16, K=16+e. Fragment storage:
// frag[w][((nt*8+kt)*32 + lane)*16 + e]  -> 32B contiguous per lane.
// ---------------------------------------------------------------------------
__global__ void prep_weights_kernel(const float* Wq, const float* Wk, const float* Wv,
                                    const float* Wo, const float* We1, __bf16* frag) {
  int idx = blockIdx.x * blockDim.x + threadIdx.x;   // 5 * 65536
  if (idx >= 5 * 65536) return;
  int widx = idx >> 16;
  int r    = idx & 65535;
  int e    = r & 15;
  int lane = (r >> 4) & 31;
  int kt   = (r >> 9) & 7;
  int nt   = (r >> 12) & 15;
  int col  = nt * 16 + (lane & 15);
  int k    = kt * 32 + ((lane >= 16) ? 16 : 0) + e;
  const float* W = (widx == 0) ? Wq : (widx == 1) ? Wk : (widx == 2) ? Wv
                 : (widx == 3) ? Wo : We1;
  frag[idx] = (__bf16)W[k * 256 + col];
}

// ---------------------------------------------------------------------------
// Kernel 2: QKV projections. 192 waves; each wave: one 16-row tile of X,
// 16 N-tiles x 8 K-steps of WMMA. Q,K stored bf16 [B,H,N,hd];
// V stored TRANSPOSED bf16 [B,H,hd,N] so the attention P@V B-fragments are
// single contiguous 32-byte loads.
// ---------------------------------------------------------------------------
__global__ void qkv_proj_kernel(const float* X, const __bf16* wfrag,
                                const float* bq, const float* bk, const float* bv,
                                __bf16* qf, __bf16* kf, __bf16* vtf) {
  int gw   = (blockIdx.x * blockDim.x + threadIdx.x) >> 5;  // 0..191
  int lane = threadIdx.x & 31;
  int w     = gw / 64;   // 0=q 1=k 2=v
  int mtile = gw % 64;
  const __bf16* wf   = wfrag + w * 65536;
  const float*  bias = (w == 0) ? bq : (w == 1) ? bk : bv;

  const float* xrow = X + (size_t)(mtile * 16 + (lane & 15)) * 256;
  v16bf A[8];
#pragma unroll
  for (int kt = 0; kt < 8; ++kt) A[kt] = load_a_f32_nt(xrow, kt * 32, lane);

  int rbase = mtile * 16 + ((lane >= 16) ? 8 : 0);
  for (int nt = 0; nt < 16; ++nt) {
    v8f acc = vzero8();
#pragma unroll
    for (int kt = 0; kt < 8; ++kt) {
      v16bf B = *(const v16bf*)(wf + (size_t)((nt * 8 + kt) * 32 + lane) * 16);
      acc = WMMA_BF16(A[kt], B, acc);
    }
    int n  = nt * 16 + (lane & 15);
    float bn = bias[n];
    int h = n >> 5, d = n & 31;
#pragma unroll
    for (int v = 0; v < 8; ++v) {
      int m = rbase + v;              // global row in [0,1024)
      int b = m >> 9, i = m & 511;
      __bf16 val = (__bf16)(acc[v] + bn);
      if (w == 2) {
        vtf[(((size_t)b * 8 + h) * 32 + d) * 512 + i] = val;   // [B,H,hd,N]
      } else {
        __bf16* dst = (w == 0) ? qf : kf;
        dst[(((size_t)b * 8 + h) * 512 + i) * 32 + d] = val;   // [B,H,N,hd]
      }
    }
  }
}

// ---------------------------------------------------------------------------
// Kernel 3: edge bias. 8 waves / block; We1 fragments (128 KB) staged once
// per block into LDS. One wave per 32 flattened rows of E [B*N*N, 256]
// (two 16-row A-fragment sets -> each LDS B-frag feeds 2 WMMAs, halving
// LDS read bandwidth per FLOP). E read once, non-temporal. Fused relu +
// dot(We2) epilogue; half-wave shfl_xor reduction matches the f32 C layout.
// ---------------------------------------------------------------------------
__global__ void edge_bias_kernel(const float* E, const __bf16* we1f, const float* be1,
                                 const float* We2, const float* be2, float* biasw) {
  extern __shared__ __bf16 WB[];                 // 65536 bf16 = 128 KB
  // cooperative stage: 8192 x 16B
  {
    const uint4* src = (const uint4*)we1f;
    uint4* dst = (uint4*)WB;
    for (int i = threadIdx.x; i < 8192; i += 256) dst[i] = src[i];
  }
  __syncthreads();

  int gw   = (blockIdx.x * blockDim.x + threadIdx.x) >> 5;  // 0..16383
  int lane = threadIdx.x & 31;
  size_t row0 = (size_t)gw * 32 + (lane & 15);
  const float* erow0 = E + row0 * 256;
  const float* erow1 = erow0 + 16 * 256;         // rows +16..+31

  v16bf A0[8], A1[8];
#pragma unroll
  for (int kt = 0; kt < 8; ++kt) {
    A0[kt] = load_a_f32_nt(erow0, kt * 32, lane);
    A1[kt] = load_a_f32_nt(erow1, kt * 32, lane);
  }

  float p0[8], p1[8];
#pragma unroll
  for (int v = 0; v < 8; ++v) { p0[v] = 0.f; p1[v] = 0.f; }

  for (int nt = 0; nt < 16; ++nt) {
    v8f acc0 = vzero8();
    v8f acc1 = vzero8();
#pragma unroll
    for (int kt = 0; kt < 8; ++kt) {
      v16bf B = *(const v16bf*)(WB + (size_t)((nt * 8 + kt) * 32 + lane) * 16);
      acc0 = WMMA_BF16(A0[kt], B, acc0);
      acc1 = WMMA_BF16(A1[kt], B, acc1);
    }
    int n = nt * 16 + (lane & 15);
    float b1 = be1[n];
    float w2 = We2[n];
#pragma unroll
    for (int v = 0; v < 8; ++v) {
      p0[v] += fmaxf(acc0[v] + b1, 0.f) * w2;   // relu + fused [256,1] proj
      p1[v] += fmaxf(acc1[v] + b1, 0.f) * w2;
    }
  }
  // reduce the 16 hidden columns held across each 16-lane half
#pragma unroll
  for (int mask = 1; mask < 16; mask <<= 1)
#pragma unroll
    for (int v = 0; v < 8; ++v) {
      p0[v] += __shfl_xor(p0[v], mask, 32);
      p1[v] += __shfl_xor(p1[v], mask, 32);
    }

  float b2 = be2[0];
  size_t rb = (size_t)gw * 32;
  if (lane == 0) {
#pragma unroll
    for (int v = 0; v < 8; ++v) {
      biasw[rb + v]      = p0[v] + b2;
      biasw[rb + 16 + v] = p1[v] + b2;
    }
  }
  if (lane == 16) {
#pragma unroll
    for (int v = 0; v < 8; ++v) {
      biasw[rb + 8 + v]  = p0[v] + b2;
      biasw[rb + 24 + v] = p1[v] + b2;
    }
  }
}

// ---------------------------------------------------------------------------
// Kernel 4: attention. One wave per (b,h,itile): 512 blocks x 32 threads.
// Scores (Q@K^T)*scale + bias -> 16x512 f32 LDS row block; wave-local
// softmax; P@V via WMMA (P from LDS in A-layout, V^T B-frags as contiguous
// 32B loads).
// ---------------------------------------------------------------------------
__global__ void attention_kernel(const __bf16* qf, const __bf16* kf, const __bf16* vtf,
                                 const float* biasw, float* ow) {
  __shared__ float S[16 * 512];                 // 32 KB per workgroup (1 wave)
  int g    = blockIdx.x;                        // 0..511
  int lane = threadIdx.x;                       // 0..31
  int b = g >> 8, rem = g & 255;
  int h = rem >> 5, itile = rem & 31;
  const __bf16* qb  = qf  + ((size_t)b * 8 + h) * 512 * 32;
  const __bf16* kb  = kf  + ((size_t)b * 8 + h) * 512 * 32;
  const __bf16* vtb = vtf + ((size_t)b * 8 + h) * 32 * 512;   // [hd][N]
  const int base  = (lane >= 16) ? 16 : 0;
  const int rhalf = (lane >= 16) ? 8 : 0;

  // Q A-fragment (hd = 32 -> single K step), held for the whole tile.
  const __bf16* qrow = qb + (size_t)(itile * 16 + (lane & 15)) * 32;
  v8bf qlo = *(const v8bf*)(qrow + base);
  v8bf qhi = *(const v8bf*)(qrow + 8 + base);
  v16bf aQ;
#pragma unroll
  for (int e = 0; e < 8; ++e) { aQ[e] = qlo[e]; aQ[8 + e] = qhi[e]; }

  const float scale = 0.17677669529663687f;     // 1/sqrt(32)

  for (int jt = 0; jt < 32; ++jt) {
    // B-frag = K^T tile: lanes 0-15 row j=jt*16+lane d=0..15; lanes 16-31 d=16..31
    const __bf16* kp = kb + (size_t)(jt * 16 + (lane & 15)) * 32 + ((lane >= 16) ? 16 : 0);
    v16bf bK = *(const v16bf*)kp;
    v8f s = WMMA_BF16(aQ, bK, vzero8());
    int col = jt * 16 + (lane & 15);
#pragma unroll
    for (int v = 0; v < 8; ++v) {
      int ri = itile * 16 + v + rhalf;
      float val = s[v] * scale + biasw[((size_t)(b * 512) + ri) * 512 + col];
      S[(v + rhalf) * 512 + col] = val;
    }
  }
  asm volatile("s_wait_dscnt 0x0" ::: "memory");

  // wave-local softmax over each of the 16 rows (512 cols, 16 per lane)
  for (int r = 0; r < 16; ++r) {
    float m = -1e30f;
#pragma unroll
    for (int t = 0; t < 16; ++t) m = fmaxf(m, S[r * 512 + lane + 32 * t]);
#pragma unroll
    for (int mask = 1; mask < 32; mask <<= 1) m = fmaxf(m, __shfl_xor(m, mask, 32));
    float sum = 0.f;
#pragma unroll
    for (int t = 0; t < 16; ++t) sum += __expf(S[r * 512 + lane + 32 * t] - m);
#pragma unroll
    for (int mask = 1; mask < 32; mask <<= 1) sum += __shfl_xor(sum, mask, 32);
    float inv = 1.0f / sum;
#pragma unroll
    for (int t = 0; t < 16; ++t) {
      int idx = r * 512 + lane + 32 * t;
      S[idx] = __expf(S[idx] - m) * inv;
    }
  }
  asm volatile("s_wait_dscnt 0x0" ::: "memory");

  // O = P @ V : K dim = 512 in 16 chunks of 32; hd = 32 in 2 col tiles.
  for (int nt = 0; nt < 2; ++nt) {
    v8f acc = vzero8();
    for (int jc = 0; jc < 16; ++jc) {
      const float* pr = S + (lane & 15) * 512 + jc * 32;
      v16bf aP;
#pragma unroll
      for (int e = 0; e < 8; ++e) {
        aP[e]     = (__bf16)pr[base + e];
        aP[8 + e] = (__bf16)pr[8 + base + e];
      }
      // V^T B-frag: lane holds column d fixed, 16 consecutive j -> one 32B load
      int d = nt * 16 + (lane & 15);
      const __bf16* vp = vtb + (size_t)d * 512 + jc * 32 + ((lane >= 16) ? 16 : 0);
      v16bf bV = *(const v16bf*)vp;
      acc = WMMA_BF16(aP, bV, acc);
    }
    int d = nt * 16 + (lane & 15);
#pragma unroll
    for (int v = 0; v < 8; ++v) {
      int ri = itile * 16 + v + rhalf;
      ow[((size_t)(b * 512) + ri) * 256 + h * 32 + d] = acc[v];
    }
  }
}

// ---------------------------------------------------------------------------
// Kernel 5: output projection out = O @ Wo + bo -> f32 d_out [B,N,D].
// ---------------------------------------------------------------------------
__global__ void out_proj_kernel(const float* OW, const __bf16* wof, const float* bo,
                                float* out) {
  int gw   = (blockIdx.x * blockDim.x + threadIdx.x) >> 5;  // 0..63
  int lane = threadIdx.x & 31;
  const float* orow = OW + (size_t)(gw * 16 + (lane & 15)) * 256;
  v16bf A[8];
#pragma unroll
  for (int kt = 0; kt < 8; ++kt) A[kt] = load_a_f32_nt(orow, kt * 32, lane);

  int rbase = gw * 16 + ((lane >= 16) ? 8 : 0);
  for (int nt = 0; nt < 16; ++nt) {
    v8f acc = vzero8();
#pragma unroll
    for (int kt = 0; kt < 8; ++kt) {
      v16bf B = *(const v16bf*)(wof + (size_t)((nt * 8 + kt) * 32 + lane) * 16);
      acc = WMMA_BF16(A[kt], B, acc);
    }
    int n = nt * 16 + (lane & 15);
    float bn = bo[n];
#pragma unroll
    for (int v = 0; v < 8; ++v)
      out[(size_t)(rbase + v) * 256 + n] = acc[v] + bn;
  }
}

// ---------------------------------------------------------------------------
extern "C" void kernel_launch(void* const* d_in, const int* in_sizes, int n_in,
                              void* d_out, int out_size, void* d_ws, size_t ws_size,
                              hipStream_t stream) {
  const float* X   = (const float*)d_in[0];
  const float* E   = (const float*)d_in[1];
  const float* Wq  = (const float*)d_in[2];
  const float* bq  = (const float*)d_in[3];
  const float* Wk  = (const float*)d_in[4];
  const float* bk  = (const float*)d_in[5];
  const float* Wv  = (const float*)d_in[6];
  const float* bv  = (const float*)d_in[7];
  const float* Wo  = (const float*)d_in[8];
  const float* bo  = (const float*)d_in[9];
  const float* We1 = (const float*)d_in[10];
  const float* be1 = (const float*)d_in[11];
  const float* We2 = (const float*)d_in[12];
  const float* be2 = (const float*)d_in[13];
  float* out = (float*)d_out;

  // Workspace layout (~5.4 MB total):
  char* ws = (char*)d_ws;
  __bf16* qf    = (__bf16*)ws;                     // 262144 bf16
  __bf16* kf    = qf + 262144;                     // 262144 bf16
  __bf16* vtf   = kf + 262144;                     // 262144 bf16 (V transposed)
  float*  biasw = (float*)(ws + 1572864);          // 524288 f32 (bias [B,N,N])
  float*  ow    = (float*)(ws + 3670016);          // 262144 f32 (O [B*N, 256])
  __bf16* wfrag = (__bf16*)(ws + 4718592);         // 5 x 65536 bf16 weight frags

  prep_weights_kernel<<<(5 * 65536) / 256, 256, 0, stream>>>(Wq, Wk, Wv, Wo, We1, wfrag);
  // 192 waves (3 weights x 64 row tiles)
  qkv_proj_kernel<<<48, 128, 0, stream>>>(X, wfrag, bq, bk, bv, qf, kf, vtf);
  // 16384 waves (32 E-rows each); 8 waves/block share 128 KB LDS We1 frags
  edge_bias_kernel<<<2048, 256, 131072, stream>>>(E, wfrag + 4 * 65536, be1, We2, be2, biasw);
  // 512 waves, one per (b, h, 16-row tile), 32 KB LDS each
  attention_kernel<<<512, 32, 0, stream>>>(qf, kf, vtf, biasw, ow);
  // 64 waves
  out_proj_kernel<<<16, 128, 0, stream>>>(ow, wfrag + 3 * 65536, bo, out);
}
